// GRU_66941360275690
// MI455X (gfx1250) — matrix operand order repaired and checked by reading, not compiled, not run on hardware
//
#include <hip/hip_runtime.h>
#include <hip/hip_bf16.h>
#include <math.h>

// GRU: T=512, B=128, IN=256, H=256, G=3H=768
#define T_STEPS 512
#define BATCH   128
#define IN_DIM  256
#define H_DIM   256
#define G_DIM   768
#define NWG     16          // persistent workgroups in the scan kernel

typedef __attribute__((ext_vector_type(16))) __bf16 v16bf;
typedef __attribute__((ext_vector_type(8)))  __bf16 v8bf;
typedef __attribute__((ext_vector_type(8)))  float  v8f;

#if defined(__has_builtin)
#if __has_builtin(__builtin_amdgcn_global_load_async_to_lds_b128) && \
    __has_builtin(__builtin_amdgcn_s_wait_asynccnt)
#define HAVE_ASYNC_LDS 1
#endif
#endif

#ifdef HAVE_ASYNC_LDS
// async copy operand types: int4 vector pointers in explicit address spaces
typedef __attribute__((__vector_size__(4 * sizeof(int)))) int v4i;
typedef __attribute__((address_space(1))) v4i gv4i;   // global
typedef __attribute__((address_space(3))) v4i lv4i;   // LDS
#endif

// ---------------------------------------------------------------------------
// WMMA fragment loaders (wave32 layouts per CDNA5 ISA 7.12.2)
// A (16x32 bf16, row-major [M][ld]): lane L -> row m0+(L&15);
//   elems 0-7  = K = k0 + (L>>4)*8 + e
//   elems 8-15 = K = k0 + 16 + (L>>4)*8 + e
// B (32x16 bf16, B(k,n) stored row-major as W[n][ld] i.e. [N][K]):
//   lane L -> col n0+(L&15); elems 0-7 = K = k0 + (L>>4)*8 + e, elems 8-15 = K = k0 + 16 + (L>>4)*8 + e
// ---------------------------------------------------------------------------
__device__ __forceinline__ v16bf frag_a(const __bf16* p, int ld, int m0, int k0) {
    int lane = threadIdx.x & 31;
    int half = lane >> 4;
    const __bf16* r = p + (size_t)(m0 + (lane & 15)) * ld + k0 + half * 8;
    v8bf lo = *(const v8bf*)(r);
    v8bf hi = *(const v8bf*)(r + 16);
    return __builtin_shufflevector(lo, hi, 0,1,2,3,4,5,6,7,8,9,10,11,12,13,14,15);
}

__device__ __forceinline__ v16bf frag_b(const __bf16* p, int ld, int n0, int k0) {
    int lane = threadIdx.x & 31;
    int half = lane >> 4;
    const __bf16* r = p + (size_t)(n0 + (lane & 15)) * ld + k0 + half * 8;
    v8bf lo = *(const v8bf*)(r);
    v8bf hi = *(const v8bf*)(r + 16);
    return __builtin_shufflevector(lo, hi, 0,1,2,3,4,5,6,7,8,9,10,11,12,13,14,15);
}

#define WMMA_BF16(a, b, c) \
    __builtin_amdgcn_wmma_f32_16x16x32_bf16(false, (a), false, (b), (short)0, (c), false, false)

// ---------------------------------------------------------------------------
// Kernel 0a: convert weights f32 -> bf16 (Wx and Wh are already [N][K] layout
// for B fragments), and reset the global barrier counter for this launch.
// ---------------------------------------------------------------------------
__global__ void k_cvt_w(const float* __restrict__ Wx, const float* __restrict__ Wh,
                        __bf16* __restrict__ wxb, __bf16* __restrict__ whb,
                        unsigned int* __restrict__ cnt) {
    if (blockIdx.x == 0 && threadIdx.x == 0) *cnt = 0u;
    int idx = blockIdx.x * blockDim.x + threadIdx.x;   // 0 .. 2*196608-1
    const int NW = G_DIM * IN_DIM;                     // 196608
    if (idx < NW)            wxb[idx]      = (__bf16)Wx[idx];
    else if (idx < 2 * NW)   whb[idx - NW] = (__bf16)Wh[idx - NW];
}

// Kernel 0b: convert x f32 -> bf16 ([T*B][IN] row-major = A-matrix layout)
__global__ void k_cvt_x(const float* __restrict__ x, __bf16* __restrict__ xb) {
    int idx = blockIdx.x * blockDim.x + threadIdx.x;
    xb[idx] = (__bf16)x[idx];
}

// ---------------------------------------------------------------------------
// Kernel 1: gx = x @ Wx^T + bx  (M=65536, K=256, N=768), bf16 WMMA, f32 acc,
// stored bf16. Per WG (8 waves): M=128 x N=64 tile; per wave: 16x64 strip.
// All operands are L2-resident (x_bf16 = 32MB < 192MB L2, Wx_bf16 = 384KB).
// ---------------------------------------------------------------------------
__global__ void k_gx_gemm(const __bf16* __restrict__ xb,    // [65536][256]
                          const __bf16* __restrict__ wxb,   // [768][256]
                          const float*  __restrict__ bx,    // [768]
                          __bf16* __restrict__ gxb) {       // [65536][768]
    const int wave = threadIdx.x >> 5;
    const int lane = threadIdx.x & 31;
    const int m0 = blockIdx.x * 128 + wave * 16;
    const int n0 = blockIdx.y * 64;

    v8f acc0 = {0.f,0.f,0.f,0.f,0.f,0.f,0.f,0.f};
    v8f acc1 = acc0, acc2 = acc0, acc3 = acc0;

#pragma unroll
    for (int k0 = 0; k0 < IN_DIM; k0 += 32) {
        v16bf a  = frag_a(xb, IN_DIM, m0, k0);
        v16bf b0 = frag_b(wxb, IN_DIM, n0 +  0, k0);
        v16bf b1 = frag_b(wxb, IN_DIM, n0 + 16, k0);
        v16bf b2 = frag_b(wxb, IN_DIM, n0 + 32, k0);
        v16bf b3 = frag_b(wxb, IN_DIM, n0 + 48, k0);
        acc0 = WMMA_BF16(a, b0, acc0);
        acc1 = WMMA_BF16(a, b1, acc1);
        acc2 = WMMA_BF16(a, b2, acc2);
        acc3 = WMMA_BF16(a, b3, acc3);
    }

    const int half = lane >> 4, nn = lane & 15;
    v8f* accs[4] = {&acc0, &acc1, &acc2, &acc3};
#pragma unroll
    for (int j = 0; j < 4; ++j) {
        int col = n0 + j * 16 + nn;
        float bias = bx[col];
        v8f a = *accs[j];
#pragma unroll
        for (int r = 0; r < 8; ++r) {
            int m = m0 + half * 8 + r;
            gxb[(size_t)m * G_DIM + col] = (__bf16)(a[r] + bias);
        }
    }
}

// ---------------------------------------------------------------------------
// Kernel 2: persistent recurrent scan. 16 WGs x 256 threads (8 waves).
// WG w owns hidden units j in [16w, 16w+16). Wave v owns batch rows
// [16v, 16v+16). The Wh B-fragments for the owned 48 gate columns are
// loop-invariant: held in 192 VGPRs for the entire 512-step scan (gfx1250 has
// a 1024-VGPR file). Per step: 24 WMMAs on the LDS-resident hidden state,
// gate math in f32, h-slice exchange through an L2 double buffer, global
// atomic barrier, then async (VGPR-free) reload of h into LDS.
// ---------------------------------------------------------------------------
__global__ void k_scan(const __bf16* __restrict__ gxb,   // [T][B][768] bf16
                       const __bf16* __restrict__ whb,   // [768][256] bf16
                       const float*  __restrict__ bh,    // [768]
                       const float*  __restrict__ h0,    // [128][256]
                       const int*    __restrict__ length,// [128]
                       __bf16* __restrict__ hbuf,        // [2][128*256] bf16
                       unsigned int* __restrict__ cnt,
                       float* __restrict__ out) {        // outputs | h_n
    __shared__ __bf16 hsh[BATCH * H_DIM];                // 64 KB: full h_t, bf16

    const int tid  = threadIdx.x;
    const int wave = tid >> 5;
    const int lane = tid & 31;
    const int half = lane >> 4, nn = lane & 15;
    const int wg = blockIdx.x;          // 0..15
    const int j0 = wg * 16;             // owned hidden-unit base
    const int m0 = wave * 16;           // owned batch-row base
    const int jj = j0 + nn;             // this lane's hidden column

    // h0 -> LDS (bf16)
    for (int i = tid; i < BATCH * H_DIM; i += 256) hsh[i] = (__bf16)h0[i];

    // Loop-invariant Wh B-fragments: [gate][kstep] -> 24 x v16bf in VGPRs
    v16bf wf[3][8];
#pragma unroll
    for (int g = 0; g < 3; ++g)
#pragma unroll
        for (int kk = 0; kk < 8; ++kk)
            wf[g][kk] = frag_b(whb, H_DIM, g * H_DIM + j0, kk * 32);

    // persistent per-wave state: owned 16x16 h slice (f32) + lengths + biases
    v8f hloc;
    int lenr[8];
#pragma unroll
    for (int r = 0; r < 8; ++r) {
        int b = m0 + half * 8 + r;
        hloc[r] = h0[b * H_DIM + jj];
        lenr[r] = length[b];
    }
    const float bhr = bh[0 * H_DIM + jj];
    const float bhz = bh[1 * H_DIM + jj];
    const float bhn = bh[2 * H_DIM + jj];
    __syncthreads();

    for (int t = 0; t < T_STEPS; ++t) {
        const __bf16* gxt = gxb + (size_t)t * BATCH * G_DIM;

        // load this step's gx pre-activations up-front (overlaps the WMMAs)
        float grv[8], gzv[8], gnv[8];
#pragma unroll
        for (int r = 0; r < 8; ++r) {
            size_t row = (size_t)(m0 + half * 8 + r) * G_DIM;
            grv[r] = (float)gxt[row + 0 * H_DIM + jj];
            gzv[r] = (float)gxt[row + 1 * H_DIM + jj];
            gnv[r] = (float)gxt[row + 2 * H_DIM + jj];
        }
        // prefetch next step's gx lines (hides HBM/L2 latency behind barrier)
        if (t + 1 < T_STEPS) {
            const __bf16* gxn = gxt + (size_t)BATCH * G_DIM;
#pragma unroll
            for (int r = 0; r < 8; ++r) {
                size_t row = (size_t)(m0 + half * 8 + r) * G_DIM;
                __builtin_prefetch(gxn + row + 0 * H_DIM + jj, 0, 0);
                __builtin_prefetch(gxn + row + 1 * H_DIM + jj, 0, 0);
                __builtin_prefetch(gxn + row + 2 * H_DIM + jj, 0, 0);
            }
        }

        // gh = h @ Wh^T for owned tiles: A from LDS, B from registers
        v8f ar = {0.f,0.f,0.f,0.f,0.f,0.f,0.f,0.f};
        v8f az = ar, an = ar;
#pragma unroll
        for (int kk = 0; kk < 8; ++kk) {
            v16bf a = frag_a(hsh, H_DIM, m0, kk * 32);
            ar = WMMA_BF16(a, wf[0][kk], ar);
            az = WMMA_BF16(a, wf[1][kk], az);
            an = WMMA_BF16(a, wf[2][kk], an);
        }

        __bf16* hnext = hbuf + (size_t)((t + 1) & 1) * (BATCH * H_DIM);
#pragma unroll
        for (int r = 0; r < 8; ++r) {
            int b = m0 + half * 8 + r;
            float rg = 1.f / (1.f + __expf(-(grv[r] + ar[r] + bhr)));
            float zg = 1.f / (1.f + __expf(-(gzv[r] + az[r] + bhz)));
            float ng = tanhf(gnv[r] + rg * (an[r] + bhn));
            float hc = (1.f - zg) * ng + zg * hloc[r];
            float hnew = (t < lenr[r]) ? hc : hloc[r];
            hloc[r] = hnew;
            out[(size_t)t * BATCH * H_DIM + (size_t)b * H_DIM + jj] = hnew;
            hnext[b * H_DIM + jj] = (__bf16)hnew;
        }

        // ---- cross-WGP step barrier (monotonic counter in L2) ----
        __threadfence();                 // publish hnext slice (device scope)
        __syncthreads();                 // all waves of this WG are done
        if (tid == 0) {
            __hip_atomic_fetch_add(cnt, 1u, __ATOMIC_RELEASE, __HIP_MEMORY_SCOPE_AGENT);
            const unsigned target = (unsigned)NWG * (unsigned)(t + 1);
            while (__hip_atomic_load(cnt, __ATOMIC_ACQUIRE, __HIP_MEMORY_SCOPE_AGENT) < target)
                __builtin_amdgcn_s_sleep(2);
        }
        __syncthreads();
        __builtin_amdgcn_fence(__ATOMIC_ACQUIRE, "agent");  // invalidate WGP$

        // pull full h_{t+1} back into LDS (64KB). Prefer the gfx1250 async
        // memory->LDS path (no VGPR staging, tracked by ASYNCcnt).
#ifdef HAVE_ASYNC_LDS
        {
            gv4i* src = (gv4i*)hnext;    // global int4*
            lv4i* dst = (lv4i*)hsh;      // LDS int4*
            for (int i = tid; i < (BATCH * H_DIM * 2) / 16; i += 256)
                __builtin_amdgcn_global_load_async_to_lds_b128(src + i, dst + i, 0, 0);
            __builtin_amdgcn_s_wait_asynccnt(0);
        }
#else
        {
            const uint4* s = (const uint4*)hnext;
            uint4*       d = (uint4*)hsh;
            for (int i = tid; i < (BATCH * H_DIM * 2) / 16; i += 256) d[i] = s[i];
        }
#endif
        __syncthreads();
    }

    // h_n = final hidden state (from persistent registers)
#pragma unroll
    for (int r = 0; r < 8; ++r) {
        int b = m0 + half * 8 + r;
        out[(size_t)T_STEPS * BATCH * H_DIM + (size_t)b * H_DIM + jj] = hloc[r];
    }
}

// ---------------------------------------------------------------------------
extern "C" void kernel_launch(void* const* d_in, const int* in_sizes, int n_in,
                              void* d_out, int out_size, void* d_ws, size_t ws_size,
                              hipStream_t stream) {
    const float* x      = (const float*)d_in[0];
    const float* h0     = (const float*)d_in[1];
    const float* Wx     = (const float*)d_in[2];
    const float* Wh     = (const float*)d_in[3];
    const float* bx     = (const float*)d_in[4];
    const float* bh     = (const float*)d_in[5];
    const int*   length = (const int*)d_in[6];
    float* out = (float*)d_out;

    // workspace layout (256B aligned)
    char* ws = (char*)d_ws;
    unsigned int* cnt = (unsigned int*)(ws + 0);
    __bf16* wxb  = (__bf16*)(ws + 256);                       // 768*256*2
    __bf16* whb  = (__bf16*)(ws + 393472);                    // 768*256*2
    __bf16* hbuf = (__bf16*)(ws + 786688);                    // 2*128*256*2
    __bf16* xb   = (__bf16*)(ws + 917760);                    // 65536*256*2
    __bf16* gxb  = (__bf16*)(ws + 34472192);                  // 65536*768*2

    k_cvt_w<<<dim3(1536), 256, 0, stream>>>(Wx, Wh, wxb, whb, cnt);
    k_cvt_x<<<dim3(65536), 256, 0, stream>>>(x, xb);
    k_gx_gemm<<<dim3(512, 12), 256, 0, stream>>>(xb, wxb, bx, gxb);
    k_scan<<<dim3(NWG), 256, 0, stream>>>(gxb, whb, bh, h0, length, hbuf, cnt, out);
}
